// MaskedGINDeepSigns_37572373906146
// MI455X (gfx1250) — matrix operand order repaired and checked by reading, not compiled, not run on hardware
//
#include <hip/hip_runtime.h>

// ---------------------------------------------------------------------------
// MaskedGINDeepSigns for MI455X (gfx1250, wave32, WMMA).
// Memory-bound: dominant cost is the per-edge K*64 gather/scatter. h and agg
// are stored in f16 (row = n*16+k, 64 cols, 2 KB per node-block): the edge
// pass is 2 KB coalesced loads + 2 KB GLOBAL_ATOMIC_PK_ADD_F16 (inline CDNA5
// asm), and h+agg = 82 MB stays L2-resident (192 MB L2).
// MLPs are fused barrier-free WMMA kernels (v_wmma_f32_16x16x32_f16, f32
// accumulate); weights pre-swizzled into B-fragment lane layout in global
// memory (2x global_load_b128 per fragment). LDS tiles use ldm=72 halfs
// (144 B rows): 16B-aligned b128 A-frag loads, conflict-free banks.
// Tail handling via a block-uniform `full` branch (no per-element exec churn).
// ---------------------------------------------------------------------------

typedef __attribute__((ext_vector_type(16))) _Float16 v16h;
typedef __attribute__((ext_vector_type(8)))  _Float16 v8h;
typedef __attribute__((ext_vector_type(4)))  _Float16 v4h;
typedef __attribute__((ext_vector_type(8)))  float    v8f;

#define WMMA_F16(a, b, c) \
  __builtin_amdgcn_wmma_f32_16x16x32_f16(false, (a), false, (b), (short)0, (c), false, false)

#define LDM 72  // padded LDS leading dim (halfs)

// Packed f16 atomic add (2 halfs per op) -- CDNA5 GLOBAL_ATOMIC_PK_ADD_F16.
__device__ __forceinline__ void atomic_pk_add_f16(_Float16* addr, unsigned int v) {
  asm volatile("global_atomic_pk_add_f16 %0, %1, off" : : "v"(addr), "v"(v) : "memory");
}

// A matrix 16x32 f16 fragment from LDS (ISA 7.12.2 wave32 layout):
// lane L holds row m = L&15; halves 0..7 = K k0+8g+0..7, halves 8..15 =
// K k0+16+8g+0..7 (g = L>>4) -> two contiguous 16 B groups, b128 loads.
__device__ __forceinline__ v16h load_A_frag(const _Float16* lds, int row0,
                                            int k0, int lane) {
  const int m = lane & 15, g = lane >> 4;
  const _Float16* p = lds + (row0 + m) * LDM + k0 + (g << 3);
  v16h a;
#pragma unroll
  for (int h = 0; h < 8; ++h) a[h] = p[h];
#pragma unroll
  for (int h = 0; h < 8; ++h) a[8 + h] = p[16 + h];
  return a;
}

// B fragment: pre-swizzled in global memory as [tile][lane][16] halfs.
__device__ __forceinline__ v16h load_B_swz(const _Float16* __restrict__ swW,
                                           int tile, int lane) {
  return *(const v16h*)(swW + ((size_t)(tile * 32 + lane) << 4));
}

// Pre-swizzle a K=64 x C weight matrix (f32, row-major) into B-fragment order.
__global__ void swz_weight_kernel(const float* __restrict__ W,
                                  _Float16* __restrict__ out, int C) {
  const int ntn = C >> 4;
  const int total = C * 64;
  for (int j = blockIdx.x * blockDim.x + threadIdx.x; j < total;
       j += gridDim.x * blockDim.x) {
    const int h = j & 15;
    const int L = (j >> 4) & 31;
    const int t = j >> 9;
    const int kt = t / ntn, nt = t - kt * ntn;
    const int k = kt * 32 + ((L >> 4) << 3) + (h & 7) + ((h >> 3) << 4);
    const int n = (nt << 4) + (L & 15);
    out[j] = (_Float16)W[k * C + n];
  }
}

// ---- utility kernels --------------------------------------------------------

__global__ void zero_f32x4_kernel(float4* __restrict__ p, long n4) {
  const float4 z = {0.f, 0.f, 0.f, 0.f};
  for (long i = blockIdx.x * (long)blockDim.x + threadIdx.x; i < n4;
       i += (long)gridDim.x * blockDim.x)
    p[i] = z;
}

__global__ void zero_i32_kernel(int* __restrict__ p, int n) {
  int i = blockIdx.x * blockDim.x + threadIdx.x;
  if (i < n) p[i] = 0;
}

__global__ void hist_kernel(const int* __restrict__ batch, int* __restrict__ cnt, int n) {
  int i = blockIdx.x * blockDim.x + threadIdx.x;
  if (i < n) atomicAdd(&cnt[batch[i]], 1);
}

__global__ void nnodes_kernel(const int* __restrict__ batch, const int* __restrict__ cnt,
                              int* __restrict__ nn, int n) {
  int i = blockIdx.x * blockDim.x + threadIdx.x;
  if (i < n) nn[i] = cnt[batch[i]];
}

// z0[n*16+k] = sign * x[n*16+k]   (layer-0 input, f32)
__global__ void init_z0_kernel(const float* __restrict__ x, float* __restrict__ z0,
                               float sign, long n) {
  long i = blockIdx.x * (long)blockDim.x + threadIdx.x;
  if (i < n) z0[i] = sign * x[i];
}

// layer-0 edge aggregate: z0[dst,k] += sign * x[src,k]  (f32 atomics, tiny)
__global__ void edge0_kernel(const float* __restrict__ x, const int* __restrict__ ei,
                             float* __restrict__ z0, int E, float sign) {
  long i = blockIdx.x * (long)blockDim.x + threadIdx.x;
  if (i >= (long)E * 16) return;
  const int e = (int)(i >> 4), k = (int)(i & 15);
  const int src = ei[e], dst = ei[E + e];
  atomicAdd(&z0[(size_t)dst * 16 + k], sign * x[(size_t)src * 16 + k]);
}

// layers 1..2 edge aggregate: agg[dst,k,c] += h[src,k,c] in f16.
// 2 edges per 256-thread block; per edge: 2 KB coalesced b128 loads +
// 512 packed-f16 atomics (global_atomic_pk_add_f16).
__global__ void edge_scatter_kernel(const _Float16* __restrict__ h,
                                    const int* __restrict__ ei,
                                    _Float16* __restrict__ agg, int E) {
  const int e = blockIdx.x * 2 + (threadIdx.x >> 7);
  if (e >= E) return;
  const int t = threadIdx.x & 127;  // 128 threads/edge, 8 halfs each
  const int src = ei[e], dst = ei[E + e];
  const uint4 v = *(const uint4*)(h + (size_t)src * 1024 + t * 8);
  _Float16* d = agg + (size_t)dst * 1024 + t * 8;
  atomic_pk_add_f16(d + 0, v.x);
  atomic_pk_add_f16(d + 2, v.y);
  atomic_pk_add_f16(d + 4, v.z);
  atomic_pk_add_f16(d + 6, v.w);
}

// masked sum over K: pooled[n,c] += sum_{k < min(16, n_nodes[n])} h[n*16+k, c]
__global__ void pool_kernel(const _Float16* __restrict__ h, const int* __restrict__ nn,
                            float* __restrict__ pooled, int N) {
  int idx = blockIdx.x * blockDim.x + threadIdx.x;
  if (idx >= N * 64) return;
  const int n = idx >> 6, c = idx & 63;
  int kmax = nn[n];
  if (kmax > 16) kmax = 16;
  float s = 0.f;
  for (int k = 0; k < kmax; ++k) s += (float)h[((size_t)n * 16 + k) * 64 + c];
  pooled[idx] += s;
}

// ---- fused MLP kernels (WMMA, barrier-free: each wave owns its 16 rows) ----

// Layer 0 (C_IN=1): t = relu(z0 * W1[0,:] + b1) scalar rank-1, then
// h = t @ W2 + b2 via WMMA -> f16 output staged through LDS, b128 stores.
__global__ __launch_bounds__(128) void mlp0_kernel(
    const float* __restrict__ z0, const float* __restrict__ W1,
    const float* __restrict__ b1, const _Float16* __restrict__ sw2,
    const float* __restrict__ b2, _Float16* __restrict__ hout, int nrows) {
  __shared__ _Float16 sT[64 * LDM];
  const int tid = threadIdx.x, lane = tid & 31, wave = tid >> 5;
  const int row0 = blockIdx.x * 64;
  const int mrow = wave * 16;
  const int n = lane & 15, g = lane >> 4;
  const bool full = (row0 + 64) <= nrows;

  if (full) {
#pragma unroll
    for (int it = 0; it < 32; ++it) {
      const int j = it * 32 + lane;
      const int r = j >> 6, c = j & 63;
      const float v = z0[row0 + mrow + r] * W1[c] + b1[c];
      sT[(mrow + r) * LDM + c] = (_Float16)(v > 0.f ? v : 0.f);
    }
  } else {
    for (int it = 0; it < 32; ++it) {
      const int j = it * 32 + lane;
      const int r = j >> 6, c = j & 63;
      const int row = row0 + mrow + r;
      const float z = (row < nrows) ? z0[row] : 0.f;
      const float v = z * W1[c] + b1[c];
      sT[(mrow + r) * LDM + c] = (_Float16)(v > 0.f ? v : 0.f);
    }
  }

  const v16h t0 = load_A_frag(sT, mrow, 0, lane);
  const v16h t1 = load_A_frag(sT, mrow, 32, lane);
#pragma unroll
  for (int nt = 0; nt < 4; ++nt) {
    v8f c8 = {0.f, 0.f, 0.f, 0.f, 0.f, 0.f, 0.f, 0.f};
    c8 = WMMA_F16(t0, load_B_swz(sw2, nt, lane), c8);
    c8 = WMMA_F16(t1, load_B_swz(sw2, 4 + nt, lane), c8);
    const float bias = b2[nt * 16 + n];
#pragma unroll
    for (int r = 0; r < 8; ++r)
      sT[(mrow + r + 8 * g) * LDM + nt * 16 + n] = (_Float16)(c8[r] + bias);
  }
  if (full) {
#pragma unroll
    for (int it = 0; it < 4; ++it) {
      const int j = it * 32 + lane;
      const int r = j >> 3, c8i = j & 7;
      *(v8h*)(hout + (size_t)(row0 + mrow + r) * 64 + c8i * 8) =
          *(const v8h*)&sT[(mrow + r) * LDM + c8i * 8];
    }
  } else {
    for (int it = 0; it < 4; ++it) {
      const int j = it * 32 + lane;
      const int r = j >> 3, c8i = j & 7;
      const int row = row0 + mrow + r;
      if (row < nrows)
        *(v8h*)(hout + (size_t)row * 64 + c8i * 8) =
            *(const v8h*)&sT[(mrow + r) * LDM + c8i * 8];
    }
  }
}

// Layers 1..2: z = hin + agg (pk f16 add), t = relu(z@W1+b1), hout = t@W2+b2.
// hin/hout may alias (each row owned by exactly one wave).
__global__ __launch_bounds__(128) void mlp64_kernel(
    const _Float16* hin, const _Float16* __restrict__ agg,
    const _Float16* __restrict__ sw1, const float* __restrict__ b1,
    const _Float16* __restrict__ sw2, const float* __restrict__ b2,
    _Float16* hout, int nrows) {
  __shared__ _Float16 sZ[64 * LDM];
  __shared__ _Float16 sT[64 * LDM];
  const int tid = threadIdx.x, lane = tid & 31, wave = tid >> 5;
  const int row0 = blockIdx.x * 64;
  const int mrow = wave * 16;
  const int n = lane & 15, g = lane >> 4;
  const bool full = (row0 + 64) <= nrows;

  // fill this wave's 16 rows of sZ: z = h + agg (v_pk_add_f16), b128 I/O
  if (full) {
#pragma unroll
    for (int it = 0; it < 4; ++it) {
      const int j = it * 32 + lane;  // 0..127 v8h slots in 16x64 tile
      const int r = j >> 3, c8i = j & 7;
      const size_t go = (size_t)(row0 + mrow + r) * 64 + c8i * 8;
      const v8h a = *(const v8h*)(hin + go);
      const v8h b = *(const v8h*)(agg + go);
      *(v8h*)&sZ[(mrow + r) * LDM + c8i * 8] = a + b;
    }
  } else {
    for (int it = 0; it < 4; ++it) {
      const int j = it * 32 + lane;
      const int r = j >> 3, c8i = j & 7;
      const int row = row0 + mrow + r;
      v8h z = {0, 0, 0, 0, 0, 0, 0, 0};
      if (row < nrows) {
        const size_t go = (size_t)row * 64 + c8i * 8;
        const v8h a = *(const v8h*)(hin + go);
        const v8h b = *(const v8h*)(agg + go);
        z = a + b;
      }
      *(v8h*)&sZ[(mrow + r) * LDM + c8i * 8] = z;
    }
  }

  // GEMM1: sZ @ W1, +b1, ReLU -> sT
  const v16h a0 = load_A_frag(sZ, mrow, 0, lane);
  const v16h a1 = load_A_frag(sZ, mrow, 32, lane);
#pragma unroll
  for (int nt = 0; nt < 4; ++nt) {
    v8f c8 = {0.f, 0.f, 0.f, 0.f, 0.f, 0.f, 0.f, 0.f};
    c8 = WMMA_F16(a0, load_B_swz(sw1, nt, lane), c8);
    c8 = WMMA_F16(a1, load_B_swz(sw1, 4 + nt, lane), c8);
    const float bias = b1[nt * 16 + n];
#pragma unroll
    for (int r = 0; r < 8; ++r) {
      const float v = c8[r] + bias;
      sT[(mrow + r + 8 * g) * LDM + nt * 16 + n] = (_Float16)(v > 0.f ? v : 0.f);
    }
  }

  // GEMM2: sT @ W2, +b2; stage f16 result back into sZ (dead), b128 stores
  const v16h t0 = load_A_frag(sT, mrow, 0, lane);
  const v16h t1 = load_A_frag(sT, mrow, 32, lane);
#pragma unroll
  for (int nt = 0; nt < 4; ++nt) {
    v8f c8 = {0.f, 0.f, 0.f, 0.f, 0.f, 0.f, 0.f, 0.f};
    c8 = WMMA_F16(t0, load_B_swz(sw2, nt, lane), c8);
    c8 = WMMA_F16(t1, load_B_swz(sw2, 4 + nt, lane), c8);
    const float bias = b2[nt * 16 + n];
#pragma unroll
    for (int r = 0; r < 8; ++r)
      sZ[(mrow + r + 8 * g) * LDM + nt * 16 + n] = (_Float16)(c8[r] + bias);
  }
  if (full) {
#pragma unroll
    for (int it = 0; it < 4; ++it) {
      const int j = it * 32 + lane;
      const int r = j >> 3, c8i = j & 7;
      *(v8h*)(hout + (size_t)(row0 + mrow + r) * 64 + c8i * 8) =
          *(const v8h*)&sZ[(mrow + r) * LDM + c8i * 8];
    }
  } else {
    for (int it = 0; it < 4; ++it) {
      const int j = it * 32 + lane;
      const int r = j >> 3, c8i = j & 7;
      const int row = row0 + mrow + r;
      if (row < nrows)
        *(v8h*)(hout + (size_t)row * 64 + c8i * 8) =
            *(const v8h*)&sZ[(mrow + r) * LDM + c8i * 8];
    }
  }
}

// rho: out = relu(pooled @ rW1 + rb1) @ rW2 + rb2   ([N,64] f32 -> [N,16] f32)
__global__ __launch_bounds__(128) void rho_kernel(
    const float* __restrict__ pooled, const _Float16* __restrict__ sw1,
    const float* __restrict__ b1, const _Float16* __restrict__ sw2,
    const float* __restrict__ b2, float* __restrict__ out, int nrows) {
  __shared__ _Float16 sZ[64 * LDM];
  __shared__ _Float16 sT[64 * LDM];
  const int tid = threadIdx.x, lane = tid & 31, wave = tid >> 5;
  const int row0 = blockIdx.x * 64;
  const int mrow = wave * 16;
  const int n = lane & 15, g = lane >> 4;
  const bool full = (row0 + 64) <= nrows;

  if (full) {
#pragma unroll
    for (int it = 0; it < 8; ++it) {
      const int j = it * 32 + lane;
      const int r = j >> 4, c4 = j & 15;
      const float4 v = *(const float4*)(pooled + (size_t)(row0 + mrow + r) * 64 + c4 * 4);
      v4h hv = {(_Float16)v.x, (_Float16)v.y, (_Float16)v.z, (_Float16)v.w};
      *(v4h*)&sZ[(mrow + r) * LDM + c4 * 4] = hv;
    }
  } else {
    for (int it = 0; it < 8; ++it) {
      const int j = it * 32 + lane;
      const int r = j >> 4, c4 = j & 15;
      const int row = row0 + mrow + r;
      float4 v = {0.f, 0.f, 0.f, 0.f};
      if (row < nrows) v = *(const float4*)(pooled + (size_t)row * 64 + c4 * 4);
      v4h hv = {(_Float16)v.x, (_Float16)v.y, (_Float16)v.z, (_Float16)v.w};
      *(v4h*)&sZ[(mrow + r) * LDM + c4 * 4] = hv;
    }
  }

  const v16h a0 = load_A_frag(sZ, mrow, 0, lane);
  const v16h a1 = load_A_frag(sZ, mrow, 32, lane);
#pragma unroll
  for (int nt = 0; nt < 4; ++nt) {
    v8f c8 = {0.f, 0.f, 0.f, 0.f, 0.f, 0.f, 0.f, 0.f};
    c8 = WMMA_F16(a0, load_B_swz(sw1, nt, lane), c8);
    c8 = WMMA_F16(a1, load_B_swz(sw1, 4 + nt, lane), c8);
    const float bias = b1[nt * 16 + n];
#pragma unroll
    for (int r = 0; r < 8; ++r) {
      const float v = c8[r] + bias;
      sT[(mrow + r + 8 * g) * LDM + nt * 16 + n] = (_Float16)(v > 0.f ? v : 0.f);
    }
  }

  // GEMM2: 16 output cols = single tile column (swizzled rW2: tiles {0,1})
  const v16h t0 = load_A_frag(sT, mrow, 0, lane);
  const v16h t1 = load_A_frag(sT, mrow, 32, lane);
  v8f c8 = {0.f, 0.f, 0.f, 0.f, 0.f, 0.f, 0.f, 0.f};
  c8 = WMMA_F16(t0, load_B_swz(sw2, 0, lane), c8);
  c8 = WMMA_F16(t1, load_B_swz(sw2, 1, lane), c8);
  const float bias = b2[n];
  if (full) {
#pragma unroll
    for (int r = 0; r < 8; ++r)
      out[(size_t)(row0 + mrow + r + 8 * g) * 16 + n] = c8[r] + bias;
  } else {
    for (int r = 0; r < 8; ++r) {
      const int rr = row0 + mrow + r + 8 * g;
      if (rr < nrows) out[(size_t)rr * 16 + n] = c8[r] + bias;
    }
  }
}

// ---------------------------------------------------------------------------

extern "C" void kernel_launch(void* const* d_in, const int* in_sizes, int n_in,
                              void* d_out, int out_size, void* d_ws, size_t ws_size,
                              hipStream_t stream) {
  (void)n_in; (void)out_size; (void)ws_size;

  const float* x     = (const float*)d_in[0];
  const int*   ei    = (const int*)d_in[1];
  const int*   batch = (const int*)d_in[2];
  const float* gW1[3] = {(const float*)d_in[3],  (const float*)d_in[7],  (const float*)d_in[11]};
  const float* gb1[3] = {(const float*)d_in[4],  (const float*)d_in[8],  (const float*)d_in[12]};
  const float* gW2[3] = {(const float*)d_in[5],  (const float*)d_in[9],  (const float*)d_in[13]};
  const float* gb2[3] = {(const float*)d_in[6],  (const float*)d_in[10], (const float*)d_in[14]};
  const float* rW1 = (const float*)d_in[15];
  const float* rb1 = (const float*)d_in[16];
  const float* rW2 = (const float*)d_in[17];
  const float* rb2 = (const float*)d_in[18];

  const int  N    = in_sizes[2];       // 20000
  const int  E    = in_sizes[1] / 2;   // 160000
  const long NK   = (long)N * 16;
  const long NK64 = NK * 64;

  // workspace layout (256 B-aligned): h(f16) | agg(f16, z0 f32 overlay) |
  // pooled(f32) | cnt | nn | swizzled weights (f16)
  char* ws = (char*)d_ws;
  size_t off = 0;
  _Float16* h   = (_Float16*)(ws + off); off += (size_t)NK64 * 2;
  _Float16* agg = (_Float16*)(ws + off); off += (size_t)NK64 * 2;
  float* pooled = (float*)(ws + off);    off += (size_t)N * 64 * 4;
  int*   cnt    = (int*)(ws + off);      off += 4096 * 4;
  int*   nn     = (int*)(ws + off);      off += ((size_t)N * 4 + 255) & ~(size_t)255;
  _Float16* swz = (_Float16*)(ws + off);
  _Float16* sw_g0W2 = swz;          // 4096 halfs each (64x64)
  _Float16* sw_g1W1 = swz + 4096;
  _Float16* sw_g1W2 = swz + 8192;
  _Float16* sw_g2W1 = swz + 12288;
  _Float16* sw_g2W2 = swz + 16384;
  _Float16* sw_rW1  = swz + 20480;
  _Float16* sw_rW2  = swz + 24576;  // 1024 halfs (64x16)
  _Float16* swW1l[3] = {nullptr, sw_g1W1, sw_g2W1};
  _Float16* swW2l[3] = {sw_g0W2, sw_g1W2, sw_g2W2};
  float* z0 = (float*)agg;          // layer-0 scalar aggregate overlays agg

  const int TB = 256;
  const int nMlpBlocks = (int)((NK + 63) / 64);

  // weight pre-swizzle into B-fragment layout (tiny, L1/L2-resident)
  swz_weight_kernel<<<8, 128, 0, stream>>>(gW2[0], sw_g0W2, 64);
  swz_weight_kernel<<<8, 128, 0, stream>>>(gW1[1], sw_g1W1, 64);
  swz_weight_kernel<<<8, 128, 0, stream>>>(gW2[1], sw_g1W2, 64);
  swz_weight_kernel<<<8, 128, 0, stream>>>(gW1[2], sw_g2W1, 64);
  swz_weight_kernel<<<8, 128, 0, stream>>>(gW2[2], sw_g2W2, 64);
  swz_weight_kernel<<<8, 128, 0, stream>>>(rW1, sw_rW1, 64);
  swz_weight_kernel<<<8, 128, 0, stream>>>(rW2, sw_rW2, 16);

  // graph sizes -> per-node n_nodes (eigenvector-slot mask)
  zero_i32_kernel<<<(4096 + TB - 1) / TB, TB, 0, stream>>>(cnt, 4096);
  zero_f32x4_kernel<<<1024, TB, 0, stream>>>((float4*)pooled, (long)N * 16);
  hist_kernel<<<(N + TB - 1) / TB, TB, 0, stream>>>(batch, cnt, N);
  nnodes_kernel<<<(N + TB - 1) / TB, TB, 0, stream>>>(batch, cnt, nn, N);

  for (int s = 0; s < 2; ++s) {
    const float sign = s ? -1.f : 1.f;

    // ---- layer 0 (C_IN=1): scalar aggregate then fused MLP ----
    init_z0_kernel<<<(int)((NK + TB - 1) / TB), TB, 0, stream>>>(x, z0, sign, NK);
    edge0_kernel<<<(int)(((long)E * 16 + TB - 1) / TB), TB, 0, stream>>>(x, ei, z0, E, sign);
    mlp0_kernel<<<nMlpBlocks, 128, 0, stream>>>(z0, gW1[0], gb1[0], sw_g0W2, gb2[0],
                                                h, (int)NK);

    // ---- layers 1..2 ----
    for (int l = 1; l < 3; ++l) {
      zero_f32x4_kernel<<<4096, TB, 0, stream>>>((float4*)agg, NK64 / 8);
      edge_scatter_kernel<<<(E + 1) / 2, 256, 0, stream>>>(h, ei, agg, E);
      mlp64_kernel<<<nMlpBlocks, 128, 0, stream>>>(h, agg, swW1l[l], gb1[l],
                                                   swW2l[l], gb2[l], h, (int)NK);
    }

    // ---- masked sum over K, accumulated across both signs ----
    pool_kernel<<<(N * 64 + TB - 1) / TB, TB, 0, stream>>>(h, nn, pooled, N);
  }

  // ---- rho MLP -> d_out [N, 16] ----
  rho_kernel<<<(N + 63) / 64, 128, 0, stream>>>(pooled, sw_rW1, rb1, sw_rW2, rb2,
                                                (float*)d_out, N);
}
